// SceneEncoderPointNetPlusPlus_61108794687612
// MI455X (gfx1250) — compile-verified, matrix operand order checked by reading, not run on hardware
//
#include <hip/hip_runtime.h>
#include <cstdint>
#include <cstddef>

typedef __attribute__((ext_vector_type(16))) __bf16 v16bf;
typedef __attribute__((ext_vector_type(8)))  float  v8f;

#define BQ   4
#define NPTS 16384

__device__ __forceinline__ unsigned short f2bf(float f) {
  unsigned u = __float_as_uint(f);
  unsigned r = u + 0x7FFFu + ((u >> 16) & 1u);   // round-to-nearest-even
  return (unsigned short)(r >> 16);
}

struct LayerDesc {
  const unsigned short* w;   // bf16 tiles: [ko][no][nn(16)][kk(32)] (B-matrix layout)
  const float*          bias;// BN-folded bias, ntiles*16 (zero padded)
  int kpad;                  // round32(c_in)
  int ntiles;                // ceil(c_out/16)
  int cout;
};

// ---------------- transpose (B,C,N) -> (B,N,C) ----------------
__global__ void transpose_kernel(const float* __restrict__ in, float* __restrict__ out,
                                 int C, int N, int total) {
  for (int t = blockIdx.x * blockDim.x + threadIdx.x; t < total;
       t += gridDim.x * blockDim.x) {
    int n = t % N;
    int c = (t / N) % C;
    int b = t / (N * C);
    out[((size_t)b * N + n) * C + c] = in[t];
  }
}

// ---------------- fold BN into weights (bf16 WMMA tiles) + bias ----------------
__global__ void prep_layer_kernel(const float* __restrict__ w, const float* __restrict__ b,
                                  const float* __restrict__ g, const float* __restrict__ bt,
                                  const float* __restrict__ mn, const float* __restrict__ vr,
                                  int cin, int cout, int ktiles, int ntiles,
                                  unsigned short* __restrict__ wout, float* __restrict__ bout) {
  const int total = ktiles * ntiles * 512;
  for (int t = blockIdx.x * blockDim.x + threadIdx.x; t < total;
       t += gridDim.x * blockDim.x) {
    const int tile = t >> 9, r = t & 511;
    const int nn = r >> 5, kk = r & 31;
    const int ko = tile / ntiles, no = tile - ko * ntiles;
    const int k = ko * 32 + kk, n = no * 16 + nn;
    float v = 0.f;
    if (k < cin && n < cout) v = w[(size_t)n * cin + k] * (g[n] * rsqrtf(vr[n] + 1e-5f));
    wout[t] = f2bf(v);
  }
  const int npad = ntiles * 16;
  for (int t = blockIdx.x * blockDim.x + threadIdx.x; t < npad;
       t += gridDim.x * blockDim.x) {
    float v = 0.f;
    if (t < cout) {
      float s = g[t] * rsqrtf(vr[t] + 1e-5f);
      v = (b[t] - mn[t]) * s + bt[t];
    }
    bout[t] = v;
  }
}

// ---------------- farthest point sampling (one block per batch) ----------------
__global__ void fps_kernel(const float* __restrict__ xyz, int sx, int n,
                           int npoint, int* __restrict__ out_idx) {
  __shared__ float dist[NPTS];
  __shared__ float rv[512];
  __shared__ int   ri[512];
  const int b = blockIdx.x, t = threadIdx.x, T = blockDim.x;
  const float* base = xyz + (size_t)b * n * sx;
  for (int i = t; i < n; i += T) dist[i] = 1e10f;
  __syncthreads();
  int farthest = 0;
  for (int it = 0; it < npoint; ++it) {
    if (t == 0) out_idx[b * npoint + it] = farthest;   // scan emits pre-update index
    const float fx = base[(size_t)farthest * sx + 0];
    const float fy = base[(size_t)farthest * sx + 1];
    const float fz = base[(size_t)farthest * sx + 2];
    float best = -1.f; int bi = n;
    for (int i = t; i < n; i += T) {
      const float* p = base + (size_t)i * sx;
      float dx = p[0] - fx, dy = p[1] - fy, dz = p[2] - fz;
      float nd = fminf(dist[i], dx * dx + dy * dy + dz * dz);
      dist[i] = nd;
      if (nd > best) { best = nd; bi = i; }
    }
    rv[t] = best; ri[t] = bi;
    __syncthreads();
    for (int off = T >> 1; off > 0; off >>= 1) {
      if (t < off) {
        float v2 = rv[t + off]; int i2 = ri[t + off];
        if (v2 > rv[t] || (v2 == rv[t] && i2 < ri[t])) { rv[t] = v2; ri[t] = i2; }
      }
      __syncthreads();
    }
    farthest = ri[0];
    __syncthreads();
  }
}

// ---------------- gather sampled centers ----------------
__global__ void gather_newxyz_kernel(const float* __restrict__ xyz, int sx, int nsrc,
                                     const int* __restrict__ idx, int npoint,
                                     float* __restrict__ out, int total) {
  for (int t = blockIdx.x * blockDim.x + threadIdx.x; t < total;
       t += gridDim.x * blockDim.x) {
    int j = t % 3;
    int s = (t / 3) % npoint;
    int b = t / (3 * npoint);
    out[t] = xyz[((size_t)b * nsrc + idx[b * npoint + s]) * sx + j];
  }
}

// ---------------- ball query: first nsample indices within radius, pad w/ first ----------------
__global__ void __launch_bounds__(32)
ball_query_kernel(const float* __restrict__ xyz, int sx, int n,
                  const float* __restrict__ new_xyz, int npoint,
                  float r2, int nsample, int* __restrict__ gidx) {
  const int s = blockIdx.x, b = blockIdx.y, lane = threadIdx.x;
  const float* base = xyz + (size_t)b * n * sx;
  const float* c = new_xyz + ((size_t)b * npoint + s) * 3;
  const float cx = c[0], cy = c[1], cz = c[2];
  int* out = gidx + ((size_t)b * npoint + s) * nsample;
  int cnt = 0, first = 0;
  for (int i0 = 0; i0 < n && cnt < nsample; i0 += 32) {
    const int i = i0 + lane;
    bool ok = false;
    if (i < n) {
      const float* p = base + (size_t)i * sx;
      float dx = p[0] - cx, dy = p[1] - cy, dz = p[2] - cz;
      ok = (dx * dx + dy * dy + dz * dz) <= r2;
    }
    unsigned mask = (unsigned)__ballot(ok);
    if (cnt == 0 && mask) first = i0 + __ffs(mask) - 1;
    if (ok) {
      int pos = cnt + __popc(mask & ((1u << lane) - 1u));
      if (pos < nsample) out[pos] = i;
    }
    cnt += __popc(mask);
  }
  if (cnt < nsample)
    for (int p = cnt + lane; p < nsample; p += 32) out[p] = first;
}

// ---------------- fused gather -> 3-layer MLP (bf16 WMMA) -> max-pool ----------------
__global__ void __launch_bounds__(32)
group_mlp_pool_kernel(const float* __restrict__ pts_src, int sp, int cin, int n_src,
                      const float* __restrict__ xyz_src, int sx,
                      const float* __restrict__ new_xyz,
                      const int* __restrict__ gidx,
                      int npoint, int nsample,
                      LayerDesc L0d, LayerDesc L1d, LayerDesc L2d,
                      float* __restrict__ out, int c_cat, int c_off, int transposed) {
  constexpr int MAXK = 288;                        // round32(259)
  __shared__ __align__(32) unsigned short buf[2][16 * MAXK]; // bf16 ping-pong
  __shared__ int   sg[16];
  __shared__ float poolsh[256];
  const int lane = threadIdx.x;
  const int s = blockIdx.x, b = blockIdx.y;
  LayerDesc L[3] = {L0d, L1d, L2d};

  for (int i = lane; i < 256; i += 32) poolsh[i] = 0.f;

  const float* ctr = new_xyz + ((size_t)b * npoint + s) * 3;
  const float cx = ctr[0], cy = ctr[1], cz = ctr[2];
  const int* gi = gidx + ((size_t)b * npoint + s) * nsample;
  const int k0pad = L[0].kpad;
  const int ctotal = cin + 3;
  const int half = lane >> 4;     // which 16-lane half of the wave
  const int lid  = lane & 15;

  const int mtiles = nsample >> 4;
  for (int mt = 0; mt < mtiles; ++mt) {
    if (lane == 0 && mt + 1 < mtiles) __builtin_prefetch(gi + (mt + 1) * 16, 0, 0);
    if (lane < 16) sg[lane] = gi[mt * 16 + lane];
    __syncthreads();
    // gather 16 samples x (cin point feats ++ 3 relative coords) as bf16
    for (int m = 0; m < 16; ++m) {
      const int g = sg[m];
      const float* pr = pts_src + ((size_t)b * n_src + g) * sp;
      const float* xr = xyz_src + ((size_t)b * n_src + g) * sx;
      for (int c = lane; c < k0pad; c += 32) {
        float v = 0.f;
        if (c < cin)          v = pr[c];
        else if (c < ctotal) {
          int j = c - cin;
          v = xr[j] - (j == 0 ? cx : (j == 1 ? cy : cz));
        }
        buf[0][m * k0pad + c] = f2bf(v);
      }
    }
    __syncthreads();

    int cur = 0;
    for (int l = 0; l < 3; ++l) {
      const LayerDesc d = L[l];
      const int outw = (l < 2) ? L[l + 1].kpad : 0;
      const int ktiles = d.kpad >> 5;
      for (int no = 0; no < d.ntiles; ++no) {
        const float bv = d.bias[no * 16 + lid];       // folded BN bias
        v8f acc = {bv, bv, bv, bv, bv, bv, bv, bv};
        for (int ko = 0; ko < ktiles; ++ko) {
          v16bf a = *reinterpret_cast<const v16bf*>(
              &buf[cur][lid * d.kpad + ko * 32 + half * 16]);
          v16bf wv = *reinterpret_cast<const v16bf*>(
              &d.w[((size_t)(ko * d.ntiles + no)) * 512 + lid * 32 + half * 16]);
          acc = __builtin_amdgcn_wmma_f32_16x16x32_bf16(
              false, a, false, wv, (short)0, acc, false, false);
        }
#pragma unroll
        for (int r = 0; r < 8; ++r) acc[r] = fmaxf(acc[r], 0.f);   // ReLU
        if (l < 2) {
          const int row0 = half << 3;  // lanes 16-31 hold rows M=8..15
#pragma unroll
          for (int r = 0; r < 8; ++r)
            buf[1 - cur][(row0 + r) * outw + no * 16 + lid] = f2bf(acc[r]);
        } else {
          // max-pool over the WMMA M dimension (samples)
          float v = acc[0];
#pragma unroll
          for (int r = 1; r < 8; ++r) v = fmaxf(v, acc[r]);
          v = fmaxf(v, __shfl_xor(v, 16, 32));       // merge the two M-halves
          if (lane < 16) poolsh[no * 16 + lane] = fmaxf(poolsh[no * 16 + lane], v);
        }
      }
      if (l < 2) {
        for (int c = d.ntiles * 16 + lane; c < outw; c += 32)
          for (int m = 0; m < 16; ++m) buf[1 - cur][m * outw + c] = 0;
        __syncthreads();
        cur = 1 - cur;
      }
    }
    __syncthreads();
  }

  if (lane < 16) {
    for (int no = 0; no < L[2].ntiles; ++no) {
      int cl = no * 16 + lane;
      if (cl < L[2].cout) {
        size_t oi = transposed
            ? (((size_t)b * c_cat + c_off + cl) * npoint + s)
            : (((size_t)b * npoint + s) * c_cat + c_off + cl);
        out[oi] = poolsh[no * 16 + lane];
      }
    }
  }
}

// =======================================================================
extern "C" void kernel_launch(void* const* d_in, const int* in_sizes, int n_in,
                              void* d_out, int out_size, void* d_ws, size_t ws_size,
                              hipStream_t stream) {
  (void)in_sizes; (void)n_in; (void)out_size; (void)ws_size;
  const float* xyz_in = (const float*)d_in[0];

  const int   nscales[3]    = {3, 2, 2};
  const int   npointS[3]    = {2048, 512, 128};
  const float radii[3][3]   = {{0.05f, 0.1f, 0.2f}, {0.2f, 0.4f, 0.f}, {0.4f, 0.8f, 0.f}};
  const int   nsampS[3][3]  = {{32, 64, 128}, {64, 128, 0}, {128, 256, 0}};
  const int   cinS[3]       = {6, 160, 256};
  const int   mlpS[3][3][3] = {{{16, 16, 32}, {32, 32, 64}, {32, 48, 64}},
                               {{64, 64, 128}, {64, 96, 128}, {0, 0, 0}},
                               {{128, 128, 256}, {128, 196, 256}, {0, 0, 0}}};
  const int   ccatS[3]      = {160, 256, 512};

  // parameter pointers: stage/scale/layer x {w,b,gamma,beta,mean,var}
  const float* P[3][3][3][6];
  int pi = 1;
  for (int st = 0; st < 3; ++st)
    for (int sc = 0; sc < nscales[st]; ++sc)
      for (int ly = 0; ly < 3; ++ly)
        for (int t = 0; t < 6; ++t) P[st][sc][ly][t] = (const float*)d_in[pi++];

  // workspace carve (deterministic layout every call)
  char* wp = (char*)d_ws;
  auto alloc = [&](size_t bytes) {
    char* p = wp; wp += (bytes + 255) & ~(size_t)255; return p;
  };
  float* pts = (float*)alloc((size_t)BQ * NPTS * 6 * 4);
  float* lp1 = (float*)alloc((size_t)BQ * 2048 * 160 * 4);
  float* lp2 = (float*)alloc((size_t)BQ * 512 * 256 * 4);
  int*   fidx[3]; float* nxyz[3];
  for (int st = 0; st < 3; ++st) {
    fidx[st] = (int*)alloc((size_t)BQ * npointS[st] * 4);
    nxyz[st] = (float*)alloc((size_t)BQ * npointS[st] * 3 * 4);
  }
  int* gidx = (int*)alloc((size_t)BQ * 2048 * 128 * 4);  // max across all scales

  LayerDesc LD[3][3][3];
  int cinL[3][3][3];
  for (int st = 0; st < 3; ++st)
    for (int sc = 0; sc < nscales[st]; ++sc)
      for (int ly = 0; ly < 3; ++ly) {
        int ci = (ly == 0) ? cinS[st] + 3 : mlpS[st][sc][ly - 1];
        int co = mlpS[st][sc][ly];
        int kpad = ((ci + 31) / 32) * 32;
        int ktiles = kpad / 32;
        int ntiles = (co + 15) / 16;
        unsigned short* wt = (unsigned short*)alloc((size_t)ktiles * ntiles * 512 * 2);
        float* bt = (float*)alloc((size_t)ntiles * 16 * 4);
        LD[st][sc][ly] = {wt, bt, kpad, ntiles, co};
        cinL[st][sc][ly] = ci;
      }

  // 1) (B,6,N) -> (B,N,6)
  {
    int total = BQ * 6 * NPTS;
    transpose_kernel<<<(total + 255) / 256, 256, 0, stream>>>(xyz_in, pts, 6, NPTS, total);
  }
  // 2) fold BN + convert weights to bf16 WMMA tiles
  for (int st = 0; st < 3; ++st)
    for (int sc = 0; sc < nscales[st]; ++sc)
      for (int ly = 0; ly < 3; ++ly) {
        const float* const* q = P[st][sc][ly];
        LayerDesc& d = LD[st][sc][ly];
        prep_layer_kernel<<<32, 256, 0, stream>>>(
            q[0], q[1], q[2], q[3], q[4], q[5],
            cinL[st][sc][ly], d.cout, d.kpad / 32, d.ntiles,
            (unsigned short*)d.w, (float*)d.bias);
      }

  // 3) the three SA-MSG stages
  for (int st = 0; st < 3; ++st) {
    const float* sxyz; int sx; const float* spts; int sp; int nsrc; int ci;
    if (st == 0)      { sxyz = pts;     sx = 6; spts = pts; sp = 6;   nsrc = NPTS; ci = 6;   }
    else if (st == 1) { sxyz = nxyz[0]; sx = 3; spts = lp1; sp = 160; nsrc = 2048; ci = 160; }
    else              { sxyz = nxyz[1]; sx = 3; spts = lp2; sp = 256; nsrc = 512;  ci = 256; }
    const int np = npointS[st];

    fps_kernel<<<BQ, 512, 0, stream>>>(sxyz, sx, nsrc, np, fidx[st]);
    {
      int total = BQ * np * 3;
      gather_newxyz_kernel<<<(total + 255) / 256, 256, 0, stream>>>(
          sxyz, sx, nsrc, fidx[st], np, nxyz[st], total);
    }

    float* outp; int transposed;
    if (st == 0)      { outp = lp1;            transposed = 0; }
    else if (st == 1) { outp = lp2;            transposed = 0; }
    else              { outp = (float*)d_out;  transposed = 1; }

    int coff = 0;
    for (int sc = 0; sc < nscales[st]; ++sc) {
      const float r  = radii[st][sc];
      const int   ns = nsampS[st][sc];
      ball_query_kernel<<<dim3(np, BQ), 32, 0, stream>>>(
          sxyz, sx, nsrc, nxyz[st], np, r * r, ns, gidx);
      group_mlp_pool_kernel<<<dim3(np, BQ), 32, 0, stream>>>(
          spts, sp, ci, nsrc, sxyz, sx, nxyz[st], gidx, np, ns,
          LD[st][sc][0], LD[st][sc][1], LD[st][sc][2],
          outp, ccatS[st], coff, transposed);
      coff += LD[st][sc][2].cout;
    }
  }
}
